// UniRepLKNetBlock_74869869904626
// MI455X (gfx1250) — matrix operand order repaired and checked by reading, not compiled, not run on hardware
//
#include <hip/hip_runtime.h>
#include <hip/hip_bf16.h>
#include <math.h>

#define N_ 2
#define C_ 128
#define H_ 128
#define W_ 128
#define Z_ 8
#define S_ 131072           // H*W*Z
#define FFN_ 512
#define NSf 262144.0f       // N*S
#define Sf 131072.0f

// ---- workspace layout (bytes) ----
#define WS_Y      0                      // float[N*C*S]  = 134217728 B
#define WS_W1B    134217728              // ushort[65536] w1 transposed [j][c] bf16
#define WS_W2TB   (134217728 + 131072)   // ushort[65536] w2 transposed [c][k] bf16
#define WS_ST     (134217728 + 262144)   // float stats region
// stat float indices
#define BRSUM 0        // 4*128
#define BRSQ  512      // 4*128
#define YSQ   1024     // 128
#define NCSUM 1152     // 2*128
#define GRNSQ 1408     // 2*512
#define PBNSUM 2432    // 128
#define PBNSQ  2560    // 128
#define ZCNT   2688    // accumulators to zero
#define BRA   2688     // 4*128
#define BRC   3200     // 4*128
#define ASC   3712     // 2*128
#define BSC   3968     // 2*128
#define GS    4224     // 2*512

typedef __attribute__((ext_vector_type(16))) __bf16 v16bf;
typedef __attribute__((ext_vector_type(8)))  float  v8f;
union BF16x16 { v16bf v; unsigned short u[16]; };

__device__ inline unsigned short f2bf(float f) {
  unsigned u = __float_as_uint(f);
  unsigned r = u + 0x7fffu + ((u >> 16) & 1u);
  return (unsigned short)(r >> 16);
}

// Branchless erf (Abramowitz-Stegun 7.1.26, |err| < 1.5e-7): rcp + fma chain + exp2.
__device__ inline float erf_appx(float x) {
  float a = fabsf(x);
  float t = __builtin_amdgcn_rcpf(fmaf(0.3275911f, a, 1.0f));
  float p = fmaf(t, 1.061405429f, -1.453152027f);
  p = fmaf(t, p, 1.421413741f);
  p = fmaf(t, p, -0.284496736f);
  p = fmaf(t, p, 0.254829592f);
  p = p * t;
  float e = __builtin_amdgcn_exp2f(-a * a * 1.4426950408889634f);
  float r = fmaf(-p, e, 1.0f);
  return copysignf(r, x);
}

__device__ inline float gelu_exact(float x) {
  return 0.5f * x * (1.0f + erf_appx(x * 0.70710678118654752f));
}

// ---- WMMA k-loop: D += A(16xK, LDS row at Arow, pair-contig) * B(Kx16, col-major pair-contig) ----
// A laundered zero *offset* (not the pointer) defeats LICM of the B-fragment
// loads across the caller's rowgroup loop (which previously caused 128-VGPR
// spills) while preserving the base pointer's global address-space provenance,
// so the loads stay global_load_b128 (not flat) and use LOADcnt only.
// w1b/w2tb are L2-resident (192MB L2): re-loading per iteration is the cheap path.
template<int KTOT>
__device__ inline v8f gemm_frag(const unsigned short* Arow, const unsigned short* Bcol,
                                int half, v8f c) {
  int off = 0;
  asm volatile("" : "+v"(off));
  const unsigned short* bp = Bcol + off;
#pragma unroll
  for (int kb = 0; kb < KTOT; kb += 32) {
    BF16x16 a, b;
#pragma unroll
    for (int v = 0; v < 8; ++v) {
      int ka = kb + ((v < 4) ? (2 * v + 8 * half) : (16 + 2 * (v - 4) + 8 * half));
      a.u[2 * v]     = Arow[ka];
      a.u[2 * v + 1] = Arow[ka + 1];
      int kx = ka;
      b.u[2 * v]     = bp[kx];
      b.u[2 * v + 1] = bp[kx + 1];
    }
    c = __builtin_amdgcn_wmma_f32_16x16x32_bf16(false, a.v, false, b.v, (short)0, c, false, false);
  }
  return c;
}

// ---- stage 16x128 bf16 A-tile to LDS: packed b32 stores, float4 global loads ----
// thread: cp = channel pair (0..63), q = row quarter (0..3)
__device__ inline void stage_tile(unsigned short* tT, const float* __restrict__ y,
                                  int nC, int sb, int cp, int q,
                                  float a0, float b0, float a1, float b1) {
  int c0 = cp * 2;
  const float4 v0 = *(const float4*)(y + (long)(nC + c0) * S_ + sb + q * 4);
  const float4 v1 = *(const float4*)(y + (long)(nC + c0 + 1) * S_ + sb + q * 4);
  const float f0[4] = {v0.x, v0.y, v0.z, v0.w};
  const float f1[4] = {v1.x, v1.y, v1.z, v1.w};
#pragma unroll
  for (int i = 0; i < 4; ++i) {
    unsigned lo = f2bf(fmaf(a0, f0[i], b0));
    unsigned hi = f2bf(fmaf(a1, f1[i], b1));
    *(unsigned*)(&tT[(q * 4 + i) * 128 + c0]) = lo | (hi << 16);
  }
}

// ---- depthwise conv branch (SAME pad, dilation RR in h/w, kz=3 pad 1) ----
template<int KK, int RR, int PP>
__device__ inline float dwb(const float* __restrict__ xp, const float* __restrict__ wgt,
                            int h, int w, int z) {
  float acc = 0.f;
#pragma unroll
  for (int kh = 0; kh < KK; ++kh) {
    int hh = h + kh * RR - PP;
    if ((unsigned)hh >= 128u) continue;
#pragma unroll
    for (int kw = 0; kw < KK; ++kw) {
      int ww = w + kw * RR - PP;
      if ((unsigned)ww >= 128u) continue;
      const float* row = xp + (((hh << 7) + ww) << 3);
      const float* wr = wgt + (kh * KK + kw) * 3;
      if (z > 0) acc = fmaf(wr[0], row[z - 1], acc);
      acc = fmaf(wr[1], row[z], acc);
      if (z < 7) acc = fmaf(wr[2], row[z + 1], acc);
    }
  }
  return acc;
}

__device__ inline float bred256(volatile float* red, float v, int tid) {
  red[tid] = v; __syncthreads();
  for (int o = 128; o > 0; o >>= 1) {
    if (tid < o) red[tid] += red[tid + o];
    __syncthreads();
  }
  float r = red[0]; __syncthreads();
  return r;
}

// ================= K0: init (zero accumulators, bf16-transpose w1/w2) =================
__global__ void k0_init(const float* __restrict__ w1, const float* __restrict__ w2,
                        unsigned short* __restrict__ w1b, unsigned short* __restrict__ w2tb,
                        float* __restrict__ st) {
  int idx = blockIdx.x * 256 + threadIdx.x;   // 65536 threads
  w1b[idx]  = f2bf(w1[(idx & 127) * FFN_ + (idx >> 7)]);   // w1b[j*128 + c]
  w2tb[idx] = f2bf(w2[(idx & 511) * C_ + (idx >> 9)]);     // w2tb[c*512 + k]
  if (idx < ZCNT) st[idx] = 0.f;
}

// ================= K1: conv branch stats =================
__global__ void k1_conv_stats(const float* __restrict__ x, const float* __restrict__ lkw,
                              const float* __restrict__ w5, const float* __restrict__ w3a,
                              const float* __restrict__ w3b, float* __restrict__ st) {
  __shared__ float wl[276];
  __shared__ float red[256];
  int tid = threadIdx.x;
  int bid = blockIdx.x;
  int chunk = bid & 15; int c = (bid >> 4) & 127; int n = bid >> 11;
  for (int i = tid; i < 276; i += 256) {
    float v;
    if (i < 147)      v = lkw[c * 147 + i];
    else if (i < 222) v = w5[c * 75 + (i - 147)];
    else if (i < 249) v = w3a[c * 27 + (i - 222)];
    else              v = w3b[c * 27 + (i - 249)];
    wl[i] = v;
  }
  __syncthreads();
  const float* xp = x + (long)(n * C_ + c) * S_;
  float s0=0,s1=0,s2=0,s3=0,q0=0,q1=0,q2=0,q3=0;
  int base = chunk * 8192;
  for (int it = 0; it < 32; ++it) {
    int s = base + it * 256 + tid;
    int z = s & 7, w = (s >> 3) & 127, h = s >> 10;
    float v0 = dwb<7,1,3>(xp, wl,       h, w, z);
    float v1 = dwb<5,1,2>(xp, wl + 147, h, w, z);
    float v2 = dwb<3,2,2>(xp, wl + 222, h, w, z);
    float v3 = dwb<3,3,3>(xp, wl + 249, h, w, z);
    s0 += v0; q0 += v0*v0; s1 += v1; q1 += v1*v1;
    s2 += v2; q2 += v2*v2; s3 += v3; q3 += v3*v3;
  }
  float r;
  r = bred256(red, s0, tid); if (tid==0) atomicAdd(&st[BRSUM + 0*128 + c], r);
  r = bred256(red, q0, tid); if (tid==0) atomicAdd(&st[BRSQ  + 0*128 + c], r);
  r = bred256(red, s1, tid); if (tid==0) atomicAdd(&st[BRSUM + 1*128 + c], r);
  r = bred256(red, q1, tid); if (tid==0) atomicAdd(&st[BRSQ  + 1*128 + c], r);
  r = bred256(red, s2, tid); if (tid==0) atomicAdd(&st[BRSUM + 2*128 + c], r);
  r = bred256(red, q2, tid); if (tid==0) atomicAdd(&st[BRSQ  + 2*128 + c], r);
  r = bred256(red, s3, tid); if (tid==0) atomicAdd(&st[BRSUM + 3*128 + c], r);
  r = bred256(red, q3, tid); if (tid==0) atomicAdd(&st[BRSQ  + 3*128 + c], r);
}

// ================= K2: fold branch BN into affine =================
__global__ void k2_braffine(const float* g0, const float* b0, const float* g1, const float* b1v,
                            const float* g2, const float* b2, const float* g3, const float* b3,
                            float* __restrict__ st) {
  int c = threadIdx.x;
  const float* gg[4] = {g0, g1, g2, g3};
  const float* bb[4] = {b0, b1v, b2, b3};
  for (int i = 0; i < 4; ++i) {
    float mean = st[BRSUM + i*128 + c] / NSf;
    float var  = st[BRSQ  + i*128 + c] / NSf - mean * mean;
    float a = gg[i][c] * rsqrtf(var + 1e-5f);
    st[BRA + i*128 + c] = a;
    st[BRC + i*128 + c] = bb[i][c] - mean * a;
  }
}

// ================= K3: fused conv sum -> y, + y stats =================
__global__ void k3_conv_fuse(const float* __restrict__ x, const float* __restrict__ lkw,
                             const float* __restrict__ w5, const float* __restrict__ w3a,
                             const float* __restrict__ w3b, float* __restrict__ st,
                             float* __restrict__ y) {
  __shared__ float wl[276];
  __shared__ float red[256];
  int tid = threadIdx.x;
  int bid = blockIdx.x;
  int chunk = bid & 15; int c = (bid >> 4) & 127; int n = bid >> 11;
  for (int i = tid; i < 276; i += 256) {
    float v;
    if (i < 147)      v = lkw[c * 147 + i];
    else if (i < 222) v = w5[c * 75 + (i - 147)];
    else if (i < 249) v = w3a[c * 27 + (i - 222)];
    else              v = w3b[c * 27 + (i - 249)];
    wl[i] = v;
  }
  __syncthreads();
  float a0 = st[BRA + c],        c0 = st[BRC + c];
  float a1 = st[BRA + 128 + c],  c1 = st[BRC + 128 + c];
  float a2 = st[BRA + 256 + c],  c2 = st[BRC + 256 + c];
  float a3 = st[BRA + 384 + c],  c3 = st[BRC + 384 + c];
  float cc = c0 + c1 + c2 + c3;
  const float* xp = x + (long)(n * C_ + c) * S_;
  float* yp = y + (long)(n * C_ + c) * S_;
  float sy = 0.f, sq = 0.f;
  int base = chunk * 8192;
  for (int it = 0; it < 32; ++it) {
    int s = base + it * 256 + tid;
    int z = s & 7, w = (s >> 3) & 127, h = s >> 10;
    float v0 = dwb<7,1,3>(xp, wl,       h, w, z);
    float v1 = dwb<5,1,2>(xp, wl + 147, h, w, z);
    float v2 = dwb<3,2,2>(xp, wl + 222, h, w, z);
    float v3 = dwb<3,3,3>(xp, wl + 249, h, w, z);
    float yv = fmaf(a0, v0, fmaf(a1, v1, fmaf(a2, v2, fmaf(a3, v3, cc))));
    yp[s] = yv;
    sy += yv; sq += yv * yv;
  }
  float r;
  r = bred256(red, sy, tid); if (tid==0) atomicAdd(&st[NCSUM + n*128 + c], r);
  r = bred256(red, sq, tid); if (tid==0) atomicAdd(&st[YSQ + c], r);
}

// ================= K4: nrm-BN affine + SE -> per-(n,c) input affine =================
__global__ void k4_se(const float* nrm_w, const float* nrm_b,
                      const float* se_dw, const float* se_db,
                      const float* se_uw, const float* se_ub,
                      float* __restrict__ st) {
  __shared__ float p[256], hsh[64], an[128], cn[128];
  int t = threadIdx.x;
  if (t < 128) {
    float sum = st[NCSUM + t] + st[NCSUM + 128 + t];
    float mean = sum / NSf;
    float var = st[YSQ + t] / NSf - mean * mean;
    float a = nrm_w[t] * rsqrtf(var + 1e-5f);
    an[t] = a; cn[t] = nrm_b[t] - mean * a;
  }
  __syncthreads();
  { int cc = t & 127;
    p[t] = an[cc] * (st[NCSUM + t] / Sf) + cn[cc]; }
  __syncthreads();
  if (t < 64) {
    int nn = t >> 5, k = t & 31;
    float acc = se_db[k];
    for (int cc = 0; cc < 128; ++cc) acc += p[nn*128 + cc] * se_dw[k*128 + cc];
    hsh[t] = fmaxf(acc, 0.f);
  }
  __syncthreads();
  { int nn = t >> 7, cc = t & 127;
    float sv = se_ub[cc];
    for (int k = 0; k < 32; ++k) sv += hsh[nn*32 + k] * se_uw[cc*32 + k];
    st[ASC + t] = an[cc] * sv;
    st[BSC + t] = cn[cc] * sv;
  }
}

// ================= K5: GEMM1 (WMMA bf16) + GRN sum-of-squares =================
__global__ void k5_gemm1_stats(const float* __restrict__ y, const unsigned short* __restrict__ w1b,
                               const float* __restrict__ b1, float* __restrict__ st) {
  __shared__ unsigned short tT[16 * 128];
  int tid = threadIdx.x, lane = tid & 31, wv = tid >> 5;
  int m = lane & 15, half = lane >> 4;
  int rowBase = blockIdx.x * 256;
  int n = rowBase / S_; int sbase0 = rowBase - n * S_;
  int cp = tid & 63, q = tid >> 6;
  float sa0 = st[ASC + n*128 + cp*2],     sb0 = st[BSC + n*128 + cp*2];
  float sa1 = st[ASC + n*128 + cp*2 + 1], sb1 = st[BSC + n*128 + cp*2 + 1];
  float acc[4] = {0.f, 0.f, 0.f, 0.f};
  for (int g = 0; g < 16; ++g) {
    int sb = sbase0 + g * 16;
    stage_tile(tT, y, n * C_, sb, cp, q, sa0, sb0, sa1, sb1);
    __syncthreads();
    const unsigned short* Arow = &tT[m * 128];
#pragma unroll
    for (int ct = 0; ct < 4; ++ct) {
      int j = (ct * 8 + wv) * 16 + m;
      v8f c = {0.f,0.f,0.f,0.f,0.f,0.f,0.f,0.f};
      c = gemm_frag<128>(Arow, &w1b[j * 128], half, c);
      float bj = b1[j];
      float s2 = 0.f;
#pragma unroll
      for (int v = 0; v < 8; ++v) {
        float gl = gelu_exact(c[v] + bj);
        s2 = fmaf(gl, gl, s2);
      }
      acc[ct] += s2;
    }
    __syncthreads();
  }
#pragma unroll
  for (int ct = 0; ct < 4; ++ct) {
    int j = (ct * 8 + wv) * 16 + m;
    atomicAdd(&st[GRNSQ + n * 512 + j], acc[ct]);
  }
}

// ================= K6: GRN scale =================
__global__ void k6_grn(const float* __restrict__ grn_g, float* __restrict__ st) {
  __shared__ float red[512];
  int j = threadIdx.x, n = blockIdx.x;
  float gx = sqrtf(st[GRNSQ + n * 512 + j]);
  red[j] = gx; __syncthreads();
  for (int o = 256; o > 0; o >>= 1) {
    if (j < o) red[j] += red[j + o];
    __syncthreads();
  }
  float mean = red[0] / 512.0f;
  st[GS + n * 512 + j] = grn_g[j] * (gx / (mean + 1e-6f)) + 1.0f;
}

// ================= K7: GEMM1 + GRN + GEMM2 (WMMA) -> r + pbn stats =================
__global__ void k7_ffn(const float* __restrict__ y, const unsigned short* __restrict__ w1b,
                       const unsigned short* __restrict__ w2tb, const float* __restrict__ b1,
                       const float* __restrict__ grn_b, float* __restrict__ st,
                       float* __restrict__ out) {
  __shared__ unsigned short tT[16 * 128];
  __shared__ unsigned short uT[16 * 512];
  int tid = threadIdx.x, lane = tid & 31, wv = tid >> 5;
  int m = lane & 15, half = lane >> 4;
  int rowBase = blockIdx.x * 256;
  int n = rowBase / S_; int sbase0 = rowBase - n * S_;
  int cp = tid & 63, q = tid >> 6;
  float sa0 = st[ASC + n*128 + cp*2],     sb0 = st[BSC + n*128 + cp*2];
  float sa1 = st[ASC + n*128 + cp*2 + 1], sb1 = st[BSC + n*128 + cp*2 + 1];
  int cOut = wv * 16 + m;
  long obase = (long)(n * C_ + cOut) * S_ + sbase0;
  float psum = 0.f, psq = 0.f;
  for (int g = 0; g < 16; ++g) {
    int sb = sbase0 + g * 16;
    stage_tile(tT, y, n * C_, sb, cp, q, sa0, sb0, sa1, sb1);
    __syncthreads();
    // GEMM1 + gelu + GRN -> uT (bf16)
    const unsigned short* Arow = &tT[m * 128];
#pragma unroll
    for (int ct = 0; ct < 4; ++ct) {
      int j = (ct * 8 + wv) * 16 + m;
      v8f c = {0.f,0.f,0.f,0.f,0.f,0.f,0.f,0.f};
      c = gemm_frag<128>(Arow, &w1b[j * 128], half, c);
      float bj = b1[j], gsj = st[GS + n * 512 + j], gbj = grn_b[j];
#pragma unroll
      for (int v = 0; v < 8; ++v) {
        float gl = gelu_exact(c[v] + bj);
        uT[(v + 8 * half) * 512 + j] = f2bf(fmaf(gsj, gl, gbj));
      }
    }
    __syncthreads();
    // GEMM2: (16x512) x (512x128), this wave -> output column cOut per lane
    const unsigned short* Arow2 = &uT[m * 512];
    v8f d = {0.f,0.f,0.f,0.f,0.f,0.f,0.f,0.f};
    d = gemm_frag<512>(Arow2, &w2tb[cOut * 512], half, d);
    long ob = obase + g * 16 + 8 * half;
    float4 o0, o1;
    o0.x = d[0]; o0.y = d[1]; o0.z = d[2]; o0.w = d[3];
    o1.x = d[4]; o1.y = d[5]; o1.z = d[6]; o1.w = d[7];
#pragma unroll
    for (int v = 0; v < 8; ++v) { psum += d[v]; psq = fmaf(d[v], d[v], psq); }
    *(float4*)(out + ob)     = o0;
    *(float4*)(out + ob + 4) = o1;
  }
  psum += __shfl_xor(psum, 16, 32);
  psq  += __shfl_xor(psq, 16, 32);
  if (half == 0) {
    atomicAdd(&st[PBNSUM + cOut], psum);
    atomicAdd(&st[PBNSQ + cOut], psq);
  }
}

// ================= K8: final pbn-BN + layer scale + residual =================
__global__ void k8_out(const float* __restrict__ x, const float* __restrict__ pbn_w,
                       const float* __restrict__ pbn_b, const float* __restrict__ ls_g,
                       const float* __restrict__ st, float* __restrict__ out) {
  int idx = blockIdx.x * 256 + threadIdx.x;
  int c = (idx >> 17) & 127;  // S = 2^17
  float mean = st[PBNSUM + c] / NSf;
  float var  = st[PBNSQ + c] / NSf - mean * mean;
  float a = pbn_w[c] * rsqrtf(var + 1e-5f);
  float r = out[idx];
  out[idx] = x[idx] + ls_g[c] * (a * (r - mean) + pbn_b[c]);
}

extern "C" void kernel_launch(void* const* d_in, const int* in_sizes, int n_in,
                              void* d_out, int out_size, void* d_ws, size_t ws_size,
                              hipStream_t stream) {
  const float* x     = (const float*)d_in[0];
  const float* lk_w  = (const float*)d_in[1];
  const float* bn0_w = (const float*)d_in[2];
  const float* bn0_b = (const float*)d_in[3];
  const float* dw5_w = (const float*)d_in[4];
  const float* dbn1_w= (const float*)d_in[5];
  const float* dbn1_b= (const float*)d_in[6];
  const float* dw3a_w= (const float*)d_in[7];
  const float* dbn2_w= (const float*)d_in[8];
  const float* dbn2_b= (const float*)d_in[9];
  const float* dw3b_w= (const float*)d_in[10];
  const float* dbn3_w= (const float*)d_in[11];
  const float* dbn3_b= (const float*)d_in[12];
  const float* nrm_w = (const float*)d_in[13];
  const float* nrm_b = (const float*)d_in[14];
  const float* se_dw = (const float*)d_in[15];
  const float* se_db = (const float*)d_in[16];
  const float* se_uw = (const float*)d_in[17];
  const float* se_ub = (const float*)d_in[18];
  const float* w1    = (const float*)d_in[19];
  const float* b1    = (const float*)d_in[20];
  const float* grn_g = (const float*)d_in[21];
  const float* grn_b = (const float*)d_in[22];
  const float* w2    = (const float*)d_in[23];
  const float* pbn_w = (const float*)d_in[24];
  const float* pbn_b = (const float*)d_in[25];
  const float* ls_g  = (const float*)d_in[26];

  char* wsb = (char*)d_ws;
  float* yb = (float*)(wsb + WS_Y);
  unsigned short* w1b  = (unsigned short*)(wsb + WS_W1B);
  unsigned short* w2tb = (unsigned short*)(wsb + WS_W2TB);
  float* st = (float*)(wsb + WS_ST);
  float* out = (float*)d_out;

  k0_init<<<256, 256, 0, stream>>>(w1, w2, w1b, w2tb, st);
  k1_conv_stats<<<N_ * C_ * 16, 256, 0, stream>>>(x, lk_w, dw5_w, dw3a_w, dw3b_w, st);
  k2_braffine<<<1, 128, 0, stream>>>(bn0_w, bn0_b, dbn1_w, dbn1_b, dbn2_w, dbn2_b, dbn3_w, dbn3_b, st);
  k3_conv_fuse<<<N_ * C_ * 16, 256, 0, stream>>>(x, lk_w, dw5_w, dw3a_w, dw3b_w, st, yb);
  k4_se<<<1, 256, 0, stream>>>(nrm_w, nrm_b, se_dw, se_db, se_uw, se_ub, st);
  k5_gemm1_stats<<<1024, 256, 0, stream>>>(yb, w1b, b1, st);
  k6_grn<<<2, 512, 0, stream>>>(grn_g, st);
  k7_ffn<<<1024, 256, 0, stream>>>(yb, w1b, w2tb, b1, grn_b, st, out);
  k8_out<<<(N_ * C_ * S_) / 256, 256, 0, stream>>>(x, pbn_w, pbn_b, ls_g, st, out);
}
